// MultiHeadAttention_66434554134744
// MI455X (gfx1250) — matrix operand order repaired and checked by reading, not compiled, not run on hardware
//
#include <hip/hip_runtime.h>

// ---------------------------------------------------------------------------
// MHA forward for MI455X (gfx1250, wave32, WMMA bf16 16x16x32, f32 accum)
// ---------------------------------------------------------------------------

typedef __bf16 bf16;
typedef __attribute__((ext_vector_type(16))) __bf16 v16bf;
typedef __attribute__((ext_vector_type(8)))  __bf16 v8bf;
typedef __attribute__((ext_vector_type(4)))  __bf16 v4bf;
typedef __attribute__((ext_vector_type(8)))  float  v8f;

constexpr int Bc  = 2;
constexpr int Sc  = 2048;
constexpr int Dc  = 1024;
constexpr int Hc  = 16;
constexpr int DKc = 64;
constexpr int Nc  = Bc * Sc;          // 4096 rows for the projection GEMMs

// ---- WMMA wrapper ---------------------------------------------------------
__device__ __forceinline__ v8f wmma_bf16(v16bf a, v16bf b, v8f c) {
  return __builtin_amdgcn_wmma_f32_16x16x32_bf16(
      false, a, false, b, (short)0, c, false, false);
}

// ---- A fragment (16x32, 16-bit), row-major source -------------------------
__device__ __forceinline__ v16bf load_a_frag(const bf16* base, int ld,
                                             int row0, int k0, int lane) {
  int r  = row0 + (lane & 15);
  int kk = k0 + ((lane & 16) ? 16 : 0);
  const bf16* p = base + (size_t)r * ld + kk;
  v8bf lo = *(const v8bf*)(p);
  v8bf hi = *(const v8bf*)(p + 8);
  return __builtin_shufflevector(lo, hi, 0, 1, 2, 3, 4, 5, 6, 7,
                                 8, 9, 10, 11, 12, 13, 14, 15);
}

// ---- B fragment (32x16, 16-bit) from a row-major [N][K] source ------------
__device__ __forceinline__ v16bf load_b_frag(const bf16* base, int ld,
                                             int col0, int k0, int lane) {
  int c  = col0 + (lane & 15);
  int kk = k0 + ((lane & 16) ? 16 : 0);
  return *(const v16bf*)(base + (size_t)c * ld + kk);
}

// ---------------------------------------------------------------------------
// f32 -> bf16 conversion (vectorized: float4 -> 4 x bf16)
// ---------------------------------------------------------------------------
__global__ void cvt_kernel(const float* __restrict__ src,
                           bf16* __restrict__ dst, int n4) {
  int i = blockIdx.x * 256 + threadIdx.x;
  if (i < n4) {
    float4 f = ((const float4*)src)[i];
    v4bf o = {(bf16)f.x, (bf16)f.y, (bf16)f.z, (bf16)f.w};
    ((v4bf*)dst)[i] = o;
  }
}

// ---------------------------------------------------------------------------
// Software-pipelined 16(M) x 64(N) wave tile: Y += X @ W^T over K = Dc.
// Next k-step fragments are loaded before the current WMMA group so loads
// overlap with matrix math (partial s_wait_loadcnt instead of drain-to-0).
// ---------------------------------------------------------------------------
__device__ __forceinline__ void gemm_tile_acc(const bf16* __restrict__ X,
                                              const bf16* __restrict__ W,
                                              int m0, int n0, int lane,
                                              v8f acc[4]) {
  v16bf a  = load_a_frag(X, Dc, m0, 0, lane);
  v16bf b0 = load_b_frag(W, Dc, n0 +  0, 0, lane);
  v16bf b1 = load_b_frag(W, Dc, n0 + 16, 0, lane);
  v16bf b2 = load_b_frag(W, Dc, n0 + 32, 0, lane);
  v16bf b3 = load_b_frag(W, Dc, n0 + 48, 0, lane);
  for (int kt = 0; kt < Dc; kt += 32) {
    int ktn = (kt + 32 < Dc) ? kt + 32 : 0;   // branchless tail (redundant ld)
    v16bf an = load_a_frag(X, Dc, m0, ktn, lane);
    v16bf c0 = load_b_frag(W, Dc, n0 +  0, ktn, lane);
    v16bf c1 = load_b_frag(W, Dc, n0 + 16, ktn, lane);
    v16bf c2 = load_b_frag(W, Dc, n0 + 32, ktn, lane);
    v16bf c3 = load_b_frag(W, Dc, n0 + 48, ktn, lane);
    acc[0] = wmma_bf16(a, b0, acc[0]);
    acc[1] = wmma_bf16(a, b1, acc[1]);
    acc[2] = wmma_bf16(a, b2, acc[2]);
    acc[3] = wmma_bf16(a, b3, acc[3]);
    a = an; b0 = c0; b1 = c1; b2 = c2; b3 = c3;
  }
}

// ---------------------------------------------------------------------------
// Projection GEMM: Y = X @ W^T + b, scatter into [B,H,S,DK] bf16 (scale for Q)
// ---------------------------------------------------------------------------
__global__ __launch_bounds__(256) void proj_kernel(
    const bf16* __restrict__ X, const bf16* __restrict__ W,
    const float* __restrict__ bias, bf16* __restrict__ Yh, float scale) {
  int lane = threadIdx.x & 31;
  int wave = threadIdx.x >> 5;
  int wt = blockIdx.x * 8 + wave;   // 4096 wave-tiles: 256 (M) x 16 (N-groups)
  int m0 = (wt >> 4) << 4;
  int n0 = (wt & 15) << 6;

  v8f acc[4] = {};
  gemm_tile_acc(X, W, m0, n0, lane, acc);

  int roff = (lane & 16) ? 8 : 0;
  int cl = lane & 15;
#pragma unroll
  for (int t = 0; t < 4; ++t) {
    int col = n0 + 16 * t + cl;
    int h = col >> 6, dk = col & (DKc - 1);
    float bcol = bias[col];
#pragma unroll
    for (int v = 0; v < 8; ++v) {
      int row = m0 + roff + v;
      int bb = row >> 11, s = row & (Sc - 1);
      float val = (acc[t][v] + bcol) * scale;
      Yh[(((size_t)bb * Hc + h) * Sc + s) * DKc + dk] = (bf16)val;
    }
  }
}

// ---------------------------------------------------------------------------
// Output projection: out = Oc @ Wo^T + bo, f32 row-major [B*S, D]
// ---------------------------------------------------------------------------
__global__ __launch_bounds__(256) void oproj_kernel(
    const bf16* __restrict__ X, const bf16* __restrict__ W,
    const float* __restrict__ bias, float* __restrict__ out) {
  int lane = threadIdx.x & 31;
  int wave = threadIdx.x >> 5;
  int wt = blockIdx.x * 8 + wave;
  int m0 = (wt >> 4) << 4;
  int n0 = (wt & 15) << 6;

  v8f acc[4] = {};
  gemm_tile_acc(X, W, m0, n0, lane, acc);

  int roff = (lane & 16) ? 8 : 0;
  int cl = lane & 15;
#pragma unroll
  for (int t = 0; t < 4; ++t) {
    int col = n0 + 16 * t + cl;
    float bcol = bias[col];
#pragma unroll
    for (int v = 0; v < 8; ++v) {
      int row = m0 + roff + v;
      out[(size_t)row * Dc + col] = acc[t][v] + bcol;
    }
  }
}

// ---------------------------------------------------------------------------
// Flash attention (causal, online softmax).
// Block = 128 threads (4 waves), 64 q-rows of one (b,h); wave owns 16 rows.
// K tile staged with GLOBAL_LOAD_ASYNC_TO_LDS_B128 (ASYNCcnt); V staged
// transposed through VGPRs; next j-tile prefetched with global_prefetch.
// ---------------------------------------------------------------------------
__global__ __launch_bounds__(128) void flash_kernel(
    const bf16* __restrict__ Qh, const bf16* __restrict__ Kh,
    const bf16* __restrict__ Vh, bf16* __restrict__ Oc) {
  __shared__ alignas(32) bf16 sK[64 * 64];        // [key][dk]
  __shared__ alignas(32) bf16 sVt[64 * 64];       // [dk][key]
  __shared__ alignas(32) bf16 sP[4][16 * 64];     // per-wave P tile [qrow][key]

  int lane = threadIdx.x & 31;
  int wave = threadIdx.x >> 5;
  int bh = blockIdx.y;                 // b*H + h
  int b = bh >> 4, h = bh & (Hc - 1);
  int Q0 = blockIdx.x * 64;
  int q0 = Q0 + wave * 16;

  const bf16* Qbase = Qh + (size_t)bh * Sc * DKc;
  const bf16* Kbase = Kh + (size_t)bh * Sc * DKc;
  const bf16* Vbase = Vh + (size_t)bh * Sc * DKc;

  v16bf aq0 = load_a_frag(Qbase, DKc, q0, 0, lane);
  v16bf aq1 = load_a_frag(Qbase, DKc, q0, 32, lane);

  v8f o[4] = {};
  float mrow[8], lrow[8];
#pragma unroll
  for (int v = 0; v < 8; ++v) { mrow[v] = -1e30f; lrow[v] = 0.f; }

  int roff = (lane & 16) ? 8 : 0;
  int cl = lane & 15;

  for (int j0 = 0; j0 <= Q0; j0 += 64) {
    __syncthreads();
    const bf16* Kb = Kbase + (size_t)j0 * DKc;
    const bf16* Vb = Vbase + (size_t)j0 * DKc;

    // --- K tile (8 KB): async global->LDS, 4 x 128 lanes x 16 B -----------
#pragma unroll
    for (int it = 0; it < 4; ++it) {
      int e0 = (it * 128 + (int)threadIdx.x) * 8;          // 8 bf16 = 16 B
      unsigned lds = (unsigned)(unsigned long long)(size_t)&sK[e0];
      unsigned long long ga = (unsigned long long)(size_t)(Kb + e0);
      asm volatile("global_load_async_to_lds_b128 %0, %1, off"
                   :: "v"(lds), "v"(ga) : "memory");
    }

    // --- V tile: b128 global reads, transpose into sVt via u16 LDS writes -
#pragma unroll
    for (int it = 0; it < 4; ++it) {
      int chunk = it * 128 + (int)threadIdx.x;             // 512 chunks of 8
      int r = chunk >> 3;                                  // key row 0..63
      int c0 = (chunk & 7) * 8;                            // dk 0..56
      v8bf vv = *(const v8bf*)(Vb + (size_t)r * DKc + c0);
#pragma unroll
      for (int u = 0; u < 8; ++u) sVt[(c0 + u) * 64 + r] = vv[u];
    }

    // --- prefetch next j-tile while this one computes ----------------------
    if (j0 < Q0) {
      const char* nk = (const char*)(Kb + 64 * DKc);
      const char* nv = (const char*)(Vb + 64 * DKc);
      __builtin_prefetch(nk + (int)threadIdx.x * 64, 0, 0);
      __builtin_prefetch(nv + (int)threadIdx.x * 64, 0, 0);
    }

    asm volatile("s_wait_asynccnt 0" ::: "memory");
    __syncthreads();

    // scores: 16 q-rows x 64 keys
    v8f sacc[4] = {};
#pragma unroll
    for (int t = 0; t < 4; ++t) {
      v16bf bk0 = load_b_frag(sK, DKc, 16 * t, 0, lane);
      sacc[t] = wmma_bf16(aq0, bk0, sacc[t]);
      v16bf bk1 = load_b_frag(sK, DKc, 16 * t, 32, lane);
      sacc[t] = wmma_bf16(aq1, bk1, sacc[t]);
    }

    // causal mask (reference: masked -> -1e9)
#pragma unroll
    for (int t = 0; t < 4; ++t) {
      int key = j0 + 16 * t + cl;
#pragma unroll
      for (int v = 0; v < 8; ++v) {
        int qg = q0 + roff + v;
        if (key > qg) sacc[t][v] = -1e9f;
      }
    }

    // online softmax; P replaces sacc
#pragma unroll
    for (int v = 0; v < 8; ++v) {
      float mx = fmaxf(fmaxf(sacc[0][v], sacc[1][v]),
                       fmaxf(sacc[2][v], sacc[3][v]));
#pragma unroll
      for (int d = 1; d < 16; d <<= 1) mx = fmaxf(mx, __shfl_xor(mx, d));
      float mnew = fmaxf(mrow[v], mx);
      float corr = __expf(mrow[v] - mnew);
      mrow[v] = mnew;
      float rs = 0.f;
#pragma unroll
      for (int t = 0; t < 4; ++t) {
        float p = __expf(sacc[t][v] - mnew);
        sacc[t][v] = p;
        rs += p;
      }
#pragma unroll
      for (int d = 1; d < 16; d <<= 1) rs += __shfl_xor(rs, d);
      lrow[v] = lrow[v] * corr + rs;
#pragma unroll
      for (int t = 0; t < 4; ++t) o[t][v] *= corr;
    }

    // transpose P from C-layout to A-layout via per-wave LDS tile
    bf16* pP = &sP[wave][0];
#pragma unroll
    for (int t = 0; t < 4; ++t)
#pragma unroll
      for (int v = 0; v < 8; ++v)
        pP[(roff + v) * 64 + 16 * t + cl] = (bf16)sacc[t][v];
    __builtin_amdgcn_wave_barrier();  // LDS in-order per wave; block reorder

    v16bf ap0 = load_a_frag(pP, 64, 0, 0, lane);
    v16bf ap1 = load_a_frag(pP, 64, 0, 32, lane);

    // O += P @ V
#pragma unroll
    for (int t = 0; t < 4; ++t) {
      v16bf bv0 = load_b_frag(sVt, 64, 16 * t, 0, lane);
      o[t] = wmma_bf16(ap0, bv0, o[t]);
      v16bf bv1 = load_b_frag(sVt, 64, 16 * t, 32, lane);
      o[t] = wmma_bf16(ap1, bv1, o[t]);
    }
  }

  // normalize and write merged-head output [B,S,D] bf16
#pragma unroll
  for (int v = 0; v < 8; ++v) {
    float inv = 1.f / lrow[v];
    int qg = q0 + roff + v;
#pragma unroll
    for (int t = 0; t < 4; ++t) {
      int dk = 16 * t + cl;
      Oc[((size_t)b * Sc + qg) * Dc + h * DKc + dk] = (bf16)(o[t][v] * inv);
    }
  }
}

// ---------------------------------------------------------------------------
// Host launch
// ---------------------------------------------------------------------------
extern "C" void kernel_launch(void* const* d_in, const int* in_sizes, int n_in,
                              void* d_out, int out_size, void* d_ws,
                              size_t ws_size, hipStream_t stream) {
  (void)in_sizes; (void)n_in; (void)out_size; (void)ws_size;

  const float* q  = (const float*)d_in[0];
  const float* k  = (const float*)d_in[1];
  const float* v  = (const float*)d_in[2];
  // d_in[3] is the causal mask; applied analytically in flash_kernel.
  const float* Wq = (const float*)d_in[4];
  const float* bq = (const float*)d_in[5];
  const float* Wk = (const float*)d_in[6];
  const float* bk = (const float*)d_in[7];
  const float* Wv = (const float*)d_in[8];
  const float* bv = (const float*)d_in[9];
  const float* Wo = (const float*)d_in[10];
  const float* bo = (const float*)d_in[11];
  float* out = (float*)d_out;

  constexpr size_t ND  = (size_t)Nc * Dc;   // 4,194,304
  constexpr size_t DSQ = (size_t)Dc * Dc;   // 1,048,576

  char* w = (char*)d_ws;
  bf16* qb  = (bf16*)w; w += ND * 2;
  bf16* kb  = (bf16*)w; w += ND * 2;
  bf16* vb  = (bf16*)w; w += ND * 2;
  bf16* Wqb = (bf16*)w; w += DSQ * 2;
  bf16* Wkb = (bf16*)w; w += DSQ * 2;
  bf16* Wvb = (bf16*)w; w += DSQ * 2;
  bf16* Wob = (bf16*)w; w += DSQ * 2;
  bf16* Qhb = (bf16*)w; w += ND * 2;
  bf16* Khb = (bf16*)w; w += ND * 2;
  bf16* Vhb = (bf16*)w; w += ND * 2;
  bf16* Ocb = (bf16*)w; w += ND * 2;   // total 64 MB

  auto cvt = [&](const float* s, bf16* d, size_t n) {
    int n4 = (int)(n / 4);
    cvt_kernel<<<(unsigned)((n4 + 255) / 256), 256, 0, stream>>>(s, d, n4);
  };
  cvt(q, qb, ND);
  cvt(k, kb, ND);
  cvt(v, vb, ND);
  cvt(Wq, Wqb, DSQ);
  cvt(Wk, Wkb, DSQ);
  cvt(Wv, Wvb, DSQ);
  cvt(Wo, Wob, DSQ);

  // 4096 wave-tiles / 8 waves per block = 512 blocks
  proj_kernel<<<512, 256, 0, stream>>>(qb, Wqb, bq, Qhb, 0.125f);  // Q/sqrt(DK)
  proj_kernel<<<512, 256, 0, stream>>>(kb, Wkb, bk, Khb, 1.0f);
  proj_kernel<<<512, 256, 0, stream>>>(vb, Wvb, bv, Vhb, 1.0f);

  dim3 ag(Sc / 64, Bc * Hc);
  flash_kernel<<<ag, 128, 0, stream>>>(Qhb, Khb, Vhb, Ocb);

  oproj_kernel<<<512, 256, 0, stream>>>(Ocb, Wob, bo, out);
}
